// SeqAttnMatch_38276748542486
// MI455X (gfx1250) — compile-verified, matrix operand order checked by reading, not run on hardware
//
#include <hip/hip_runtime.h>

// ---------------------------------------------------------------------------
// CDNA5 (gfx1250) WMMA bf16 pipeline for masked relu-FC attention.
// All GEMMs run on v_wmma_f32_16x16x32_bf16 (f32 accumulate).
// 64x64 wave tiles, double-buffered LDS, async global->LDS copies when the
// toolchain exposes the gfx1250 async builtins (ASYNCcnt path).
// ---------------------------------------------------------------------------

#define BLK 256          // 8 wave32 waves
#define BM  256          // block tile M (4 waves x 64)
#define BN  128          // block tile N (2 waves x 64)
#define BK  32
#define APITCH (BK + 8)  // 40 halfs = 80B row pitch: 16B aligned, bank-spread
#define BPITCH (BK + 8)

#if defined(__HIP_DEVICE_COMPILE__) && \
    __has_builtin(__builtin_amdgcn_global_load_async_to_lds_b128) && \
    __has_builtin(__builtin_amdgcn_s_wait_asynccnt)
#define USE_ASYNC_LDS 1
#else
#define USE_ASYNC_LDS 0
#endif

typedef __attribute__((ext_vector_type(16))) __bf16       v16bf;
typedef __attribute__((ext_vector_type(8)))  float        v8f;
typedef __attribute__((ext_vector_type(4)))  unsigned int u32x4;
typedef __attribute__((ext_vector_type(4)))  int          i32x4;

union FragU { u32x4 u[2]; v16bf v; };

__device__ __forceinline__ unsigned short f2bf(float f) {
  unsigned int u = __builtin_bit_cast(unsigned int, f);
  u += 0x7fffu + ((u >> 16) & 1u);            // round-to-nearest-even
  return (unsigned short)(u >> 16);
}

#if USE_ASYNC_LDS
typedef __attribute__((address_space(1))) i32x4* gv4_p;   // global v4i32*
typedef __attribute__((address_space(3))) i32x4* lv4_p;   // LDS    v4i32*
__device__ __forceinline__ void async_cp16(const unsigned short* g,
                                           unsigned short* l) {
  __builtin_amdgcn_global_load_async_to_lds_b128((gv4_p)g, (lv4_p)l, 0, 0);
}
#endif

// Load one 16x32 bf16 fragment (A or B operand) from LDS.
// Lane < 16 : row = lane,     K in {0..7, 16..23}
// Lane >= 16: row = lane-16,  K in {8..15, 24..31}
__device__ __forceinline__ v16bf load_frag(const unsigned short* lds,
                                           int row, int pitch, int kb) {
  FragU f;
  const unsigned short* p = lds + row * pitch + kb;
  f.u[0] = *(const u32x4*)(p);        // K kb..kb+7
  f.u[1] = *(const u32x4*)(p + 16);   // K kb+16..kb+23
  return f.v;
}

// ---------------------------------------------------------------------------
// Generic bf16 GEMM: C[m,n] = sum_k A[m,k] * B'[k,n]
//   A: [M,K] K-contiguous bf16 rows, row stride lda (halfs)
//   TRB=false: B given as [N,K] K-contiguous (row stride ldb)  (B' = rows^T)
//   TRB=true : B given as [K,N] N-contiguous (row stride ldb); transposed in LDS
// EPI 0: out bf16 = relu(acc + bias[n])
// EPI 1: out f32  = mask[n] ? -inf : acc
// EPI 2: out f32  = acc
// ---------------------------------------------------------------------------
template <int EPI, bool TRB>
__global__ __launch_bounds__(BLK)
void gemm_bf16(const unsigned short* __restrict__ A, int lda,
               const unsigned short* __restrict__ Bm, int ldb,
               const float* __restrict__ bias,
               const unsigned char* __restrict__ mask,
               void* __restrict__ Cout, int ldc, int K,
               long long aBatch, long long bBatch, long long cBatch,
               int maskBatch)
{
  __shared__ unsigned short ldsA[2][BM * APITCH];   // 2 x 20KB
  __shared__ unsigned short ldsB[2][BN * BPITCH];   // 2 x 10KB

  const int tid  = threadIdx.x;
  const int wid  = tid >> 5;
  const int lane = tid & 31;
  const int bz   = blockIdx.z;

  A  += (size_t)bz * aBatch;
  Bm += (size_t)bz * bBatch;

  const int tileM    = blockIdx.y * BM;
  const int tileN    = blockIdx.x * BN;
  const int waveMoff = (wid & 3) * 64;   // 4 waves along M (64 rows each)
  const int waveNoff = (wid >> 2) * 64;  // 2 waves along N (64 cols each)
  const int lrow     = lane & 15;
  const int kb       = (lane >> 4) * 8;  // per-lane K base inside fragment

  // ---- staging helpers -----------------------------------------------------
  // A tile: BM x BK halfs = 1024 16B segments, 4 per thread
  u32x4 pa[4];
  auto loadA = [&](int k0) {
    #pragma unroll
    for (int r = 0; r < 4; ++r) {
      int i = tid + r * BLK, row = i >> 2, seg = i & 3;
      pa[r] = *(const u32x4*)(A + (size_t)(tileM + row) * lda + k0 + seg * 8);
    }
  };
  auto storeA = [&](int buf) {
    #pragma unroll
    for (int r = 0; r < 4; ++r) {
      int i = tid + r * BLK, row = i >> 2, seg = i & 3;
      *(u32x4*)&ldsA[buf][row * APITCH + seg * 8] = pa[r];
    }
  };
  // B tile (K-contiguous source): BN x BK halfs = 512 segments, 2 per thread
  u32x4 pb[2];
  auto loadB = [&](int k0) {
    #pragma unroll
    for (int r = 0; r < 2; ++r) {
      int i = tid + r * BLK, row = i >> 2, seg = i & 3;
      pb[r] = *(const u32x4*)(Bm + (size_t)(tileN + row) * ldb + k0 + seg * 8);
    }
  };
  auto storeB = [&](int buf) {
    #pragma unroll
    for (int r = 0; r < 2; ++r) {
      int i = tid + r * BLK, row = i >> 2, seg = i & 3;
      *(u32x4*)&ldsB[buf][row * BPITCH + seg * 8] = pb[r];
    }
  };
  // B tile (N-contiguous source, transposed into LDS [n][k]): 8 dwords/thread
  unsigned int pbT[8];
  auto loadBT = [&](int k0) {
    #pragma unroll
    for (int r = 0; r < 8; ++r) {
      int i = tid + r * BLK;
      int k = i >> 6, n2 = (i & 63) * 2;
      pbT[r] = *(const unsigned int*)(Bm + (size_t)(k0 + k) * ldb + tileN + n2);
    }
  };
  auto storeBT = [&](int buf) {
    #pragma unroll
    for (int r = 0; r < 8; ++r) {
      int i = tid + r * BLK;
      int k = i >> 6, n2 = (i & 63) * 2;
      ldsB[buf][(n2 + 0) * BPITCH + k] = (unsigned short)(pbT[r] & 0xffffu);
      ldsB[buf][(n2 + 1) * BPITCH + k] = (unsigned short)(pbT[r] >> 16);
    }
  };
#if USE_ASYNC_LDS
  auto asyncA = [&](int buf, int k0) {
    #pragma unroll
    for (int r = 0; r < 4; ++r) {
      int i = tid + r * BLK, row = i >> 2, seg = i & 3;
      async_cp16(A + (size_t)(tileM + row) * lda + k0 + seg * 8,
                 &ldsA[buf][row * APITCH + seg * 8]);
    }
  };
  auto asyncB = [&](int buf, int k0) {
    #pragma unroll
    for (int r = 0; r < 2; ++r) {
      int i = tid + r * BLK, row = i >> 2, seg = i & 3;
      async_cp16(Bm + (size_t)(tileN + row) * ldb + k0 + seg * 8,
                 &ldsB[buf][row * BPITCH + seg * 8]);
    }
  };
#endif

  // ---- accumulators --------------------------------------------------------
  v8f acc[4][4];
  #pragma unroll
  for (int mi = 0; mi < 4; ++mi)
    #pragma unroll
    for (int ni = 0; ni < 4; ++ni)
      #pragma unroll
      for (int g = 0; g < 8; ++g) acc[mi][ni][g] = 0.0f;

  auto compute = [&](int buf) {
    v16bf af[4], bfr[4];
    #pragma unroll
    for (int mi = 0; mi < 4; ++mi)
      af[mi] = load_frag(&ldsA[buf][0], waveMoff + mi * 16 + lrow, APITCH, kb);
    #pragma unroll
    for (int ni = 0; ni < 4; ++ni)
      bfr[ni] = load_frag(&ldsB[buf][0], waveNoff + ni * 16 + lrow, BPITCH, kb);
    #pragma unroll
    for (int mi = 0; mi < 4; ++mi)
      #pragma unroll
      for (int ni = 0; ni < 4; ++ni)
        acc[mi][ni] = __builtin_amdgcn_wmma_f32_16x16x32_bf16(
            false, af[mi], false, bfr[ni], (short)0, acc[mi][ni], false, false);
  };

  // ---- prologue: fill buffer 0 --------------------------------------------
#if USE_ASYNC_LDS
  asyncA(0, 0);
  if (!TRB) asyncB(0, 0);
  else      { loadBT(0); storeBT(0); }
  __builtin_amdgcn_s_wait_asynccnt(0);
#else
  loadA(0); storeA(0);
  if (!TRB) { loadB(0); storeB(0); }
  else      { loadBT(0); storeBT(0); }
#endif
  __syncthreads();

  // ---- main loop: double-buffered, one barrier per K-step ------------------
  int cur = 0;
  for (int k0 = 0; k0 < K; k0 += BK) {
    const bool more = (k0 + BK) < K;
    if (more) {
#if USE_ASYNC_LDS
      asyncA(cur ^ 1, k0 + BK);
      if (!TRB) asyncB(cur ^ 1, k0 + BK);
      else      loadBT(k0 + BK);
#else
      loadA(k0 + BK);
      if (!TRB) loadB(k0 + BK);
      else      loadBT(k0 + BK);
#endif
    }
    compute(cur);
    if (more) {
#if USE_ASYNC_LDS
      if (TRB) storeBT(cur ^ 1);
      __builtin_amdgcn_s_wait_asynccnt(0);
#else
      storeA(cur ^ 1);
      if (!TRB) storeB(cur ^ 1);
      else      storeBT(cur ^ 1);
#endif
      __syncthreads();
      cur ^= 1;
    }
  }

  // ---- epilogue: C/D layout -> lane = column, vgpr g = row (+8 upper half) --
  const int half = lane >> 4;
  #pragma unroll
  for (int mi = 0; mi < 4; ++mi) {
    #pragma unroll
    for (int ni = 0; ni < 4; ++ni) {
      const int gn  = tileN + waveNoff + ni * 16 + lrow;
      const int gm0 = tileM + waveMoff + mi * 16 + half * 8;
      if (EPI == 0) {
        const float bv = bias[gn];
        unsigned short* O = (unsigned short*)Cout + (size_t)bz * cBatch;
        #pragma unroll
        for (int g = 0; g < 8; ++g) {
          float v = acc[mi][ni][g] + bv;
          v = v > 0.0f ? v : 0.0f;
          O[(size_t)(gm0 + g) * ldc + gn] = f2bf(v);
        }
      } else if (EPI == 1) {
        const bool mk = mask[(size_t)bz * maskBatch + gn] != 0;
        float* O = (float*)Cout + (size_t)bz * cBatch;
        #pragma unroll
        for (int g = 0; g < 8; ++g)
          O[(size_t)(gm0 + g) * ldc + gn] = mk ? -INFINITY : acc[mi][ni][g];
      } else {
        float* O = (float*)Cout + (size_t)bz * cBatch;
        #pragma unroll
        for (int g = 0; g < 8; ++g)
          O[(size_t)(gm0 + g) * ldc + gn] = acc[mi][ni][g];
      }
    }
  }
}

// ---------------------------------------------------------------------------
// f32 -> bf16 conversion, 4 elements / thread (b128 in, b64 out)
// ---------------------------------------------------------------------------
__global__ __launch_bounds__(256)
void cvt_bf16(const float* __restrict__ src, unsigned short* __restrict__ dst,
              int n4)
{
  int i = blockIdx.x * 256 + threadIdx.x;
  if (i >= n4) return;
  float4 v = ((const float4*)src)[i];
  uint2 o;
  o.x = (unsigned int)f2bf(v.x) | ((unsigned int)f2bf(v.y) << 16);
  o.y = (unsigned int)f2bf(v.z) | ((unsigned int)f2bf(v.w) << 16);
  *(uint2*)(dst + (size_t)i * 4) = o;
}

// ---------------------------------------------------------------------------
// Row softmax over L2=2048 masked scores (mask already baked in as -inf).
// Writes normalized alpha as bf16 IN PLACE at the start of each f32 row
// (entire row is loaded before any store -> aliasing safe).
// ---------------------------------------------------------------------------
__global__ __launch_bounds__(256)
void softmax_rows(float* __restrict__ S, int L2c)
{
  __shared__ float red[256];
  const int tid = threadIdx.x;
  float* row = S + (size_t)blockIdx.x * L2c;

  float v[8];
  float m = -INFINITY;
  #pragma unroll
  for (int j = 0; j < 8; ++j) { v[j] = row[tid + j * 256]; m = fmaxf(m, v[j]); }
  red[tid] = m; __syncthreads();
  #pragma unroll
  for (int s = 128; s > 0; s >>= 1) {
    if (tid < s) red[tid] = fmaxf(red[tid], red[tid + s]);
    __syncthreads();
  }
  m = red[0]; __syncthreads();

  float sum = 0.0f;
  #pragma unroll
  for (int j = 0; j < 8; ++j) { v[j] = __expf(v[j] - m); sum += v[j]; }
  red[tid] = sum; __syncthreads();
  #pragma unroll
  for (int s = 128; s > 0; s >>= 1) {
    if (tid < s) red[tid] += red[tid + s];
    __syncthreads();
  }
  const float inv = 1.0f / red[0];

  unsigned short* arow = (unsigned short*)row;   // bf16 alpha at same row base
  #pragma unroll
  for (int j = 0; j < 8; ++j) arow[tid + j * 256] = f2bf(v[j] * inv);
}

// ---------------------------------------------------------------------------
extern "C" void kernel_launch(void* const* d_in, const int* in_sizes, int n_in,
                              void* d_out, int out_size, void* d_ws, size_t ws_size,
                              hipStream_t stream)
{
  const float*         x     = (const float*)d_in[0];
  const float*         y     = (const float*)d_in[1];
  const unsigned char* ymask = (const unsigned char*)d_in[2];  // jax bool = 1B
  const float*         W     = (const float*)d_in[3];
  const float*         bias  = (const float*)d_in[4];
  float*               out   = (float*)d_out;

  const int Bn = 16, L1 = 2048, L2 = 2048, H = 1024;

  // workspace layout (halfs unless noted)
  unsigned short* xb = (unsigned short*)d_ws;              // x   bf16  64MB
  unsigned short* yb = xb + (size_t)Bn * L1 * H;           // y   bf16  64MB
  unsigned short* wb = yb + (size_t)Bn * L2 * H;           // W   bf16   2MB
  unsigned short* xp = wb + (size_t)H * H;                 // x_proj     64MB
  unsigned short* yp = xp + (size_t)Bn * L1 * H;           // y_proj     64MB
  float*          S  = (float*)(yp + (size_t)Bn * L2 * H); // scores f32 256MB

  // 1) convert inputs to bf16
  const int nxy4 = (Bn * L1 * H) / 4;
  const int nw4  = (H * H) / 4;
  cvt_bf16<<<(nxy4 + 255) / 256, 256, 0, stream>>>(x, xb, nxy4);
  cvt_bf16<<<(nxy4 + 255) / 256, 256, 0, stream>>>(y, yb, nxy4);
  cvt_bf16<<<(nw4 + 255) / 256, 256, 0, stream>>>(W, wb, nw4);

  // 2) projections: [32768,1024] = relu([32768,1024] @ W^T + b), W is [N,K]
  dim3 gp(H / BN, (Bn * L1) / BM, 1);
  gemm_bf16<0, false><<<gp, BLK, 0, stream>>>(xb, H, wb, H, bias, nullptr,
                                              xp, H, H, 0, 0, 0, 0);
  gemm_bf16<0, false><<<gp, BLK, 0, stream>>>(yb, H, wb, H, bias, nullptr,
                                              yp, H, H, 0, 0, 0, 0);

  // 3) scores per batch: S = x_proj @ y_proj^T, mask -> -inf
  dim3 gs(L2 / BN, L1 / BM, Bn);
  gemm_bf16<1, false><<<gs, BLK, 0, stream>>>(
      xp, H, yp, H, nullptr, ymask, S, L2, H,
      (long long)L1 * H, (long long)L2 * H, (long long)L1 * L2, L2);

  // 4) softmax rows; alpha written bf16 in place (row byte stride = L2*4)
  softmax_rows<<<Bn * L1, 256, 0, stream>>>(S, L2);

  // 5) out = alpha @ y  (A row stride L2*2 halfs; B=[K,N] transposed in LDS)
  dim3 go(H / BN, L1 / BM, Bn);
  gemm_bf16<2, true><<<go, BLK, 0, stream>>>(
      (const unsigned short*)S, L2 * 2, yb, H, nullptr, nullptr, out, H, L2,
      (long long)L1 * (L2 * 2), (long long)L2 * H, (long long)L1 * H, 0);
}
